// MultiHeadAttentionBlock_16776142258252
// MI455X (gfx1250) — matrix operand order repaired and checked by reading, not compiled, not run on hardware
//
#include <hip/hip_runtime.h>
#include <hip/hip_bf16.h>

typedef __bf16 bf16;
typedef __attribute__((ext_vector_type(16))) __bf16 v16bf;
typedef __attribute__((ext_vector_type(8)))  __bf16 v8bf;
typedef __attribute__((ext_vector_type(8)))  float  v8f;
typedef __attribute__((ext_vector_type(4)))  float  v4f;

#define DIM 4096
#define SEQ 2048
#define NH  32
#define HD  128

union FragBF { v16bf v; v8bf h[2]; };

// ---------------------------------------------------------------------------
// WMMA fragment loaders (CDNA5 16-bit layouts, CDNA5 ISA §7.12.2)
//
// A (16x32, MxK): lanes 0-15 row M=lane, halves 0..7 = K 0..7, 8..15 = K 16..23
//                 lanes 16-31 same rows, K+8 / K+24.
// B (32x16, KxN): lane (mod 16) = column N, same K order as A:
//                 lanes 0-15 K 0..7,16..23; lanes 16-31 K 8..15,24..31.
// For y = x @ W^T both A rows (x) and B columns (W rows) are contiguous in K.
// ---------------------------------------------------------------------------
__device__ inline v16bf load_frag_a(const bf16* __restrict__ p, int row0, int ld,
                                    int k0, int lane) {
  const int r = row0 + (lane & 15);
  const int k = k0 + ((lane >> 4) << 3);      // 0 or 8
  const bf16* q = p + (size_t)r * ld + k;
  FragBF f;
  f.h[0] = *(const v8bf*)(q);                 // K k .. k+7
  f.h[1] = *(const v8bf*)(q + 16);            // K k+16 .. k+23
  return f.v;
}

// Same fragment, 32-bit LDS addressing (p points into __shared__).
__device__ inline v16bf load_frag_a_lds(const bf16* p, int row0, int ld,
                                        int k0, int lane) {
  const int r = row0 + (lane & 15);
  const int k = k0 + ((lane >> 4) << 3);
  const bf16* q = p + r * ld + k;
  FragBF f;
  f.h[0] = *(const v8bf*)(q);
  f.h[1] = *(const v8bf*)(q + 16);
  return f.v;
}

__device__ inline v16bf load_frag_b(const bf16* __restrict__ p, int col0, int ld,
                                    int k0, int lane) {
  const int c = col0 + (lane & 15);
  const int k = k0 + ((lane >> 4) << 3);      // 0 or 8
  const bf16* q = p + (size_t)c * ld + k;
  FragBF f;
  f.h[0] = *(const v8bf*)(q);
  f.h[1] = *(const v8bf*)(q + 16);
  return f.v;
}

// B fragment straight from an fp32 weight matrix (inline fp32->bf16 convert;
// cheaper in total HBM bytes than a separate convert pass over 4x64MB).
__device__ inline v16bf load_frag_b_f32(const float* __restrict__ p, int col0,
                                        int ld, int k0, int lane) {
  const int c = col0 + (lane & 15);
  const int k = k0 + ((lane >> 4) << 3);
  const float* q = p + (size_t)c * ld + k;
  v16bf f;
#pragma unroll
  for (int i = 0; i < 16; i += 4) {
    v4f t = *(const v4f*)(q + (i < 8 ? i : i + 8));
    f[i + 0] = (bf16)t[0];
    f[i + 1] = (bf16)t[1];
    f[i + 2] = (bf16)t[2];
    f[i + 3] = (bf16)t[3];
  }
  return f;
}

__device__ inline v8f wmma_bf16(v16bf a, v16bf b, v8f c) {
  return __builtin_amdgcn_wmma_f32_16x16x32_bf16(false, a, false, b,
                                                 (short)0, c, false, false);
}

// ---------------------------------------------------------------------------
// fp32 -> bf16 elementwise convert
// ---------------------------------------------------------------------------
__global__ __launch_bounds__(256) void f32_to_bf16_kernel(
    const float* __restrict__ in, bf16* __restrict__ out, int n) {
  for (int i = blockIdx.x * blockDim.x + threadIdx.x; i < n;
       i += gridDim.x * blockDim.x)
    out[i] = (bf16)in[i];
}

// ---------------------------------------------------------------------------
// NT GEMM: C[M,N] = A[M,K](bf16) @ B[N,K]^T (fp32, converted inline)
//   MODE 0: write fp32 C[M,N]
//   MODE 1: write bf16 C[M,N]
//   MODE 2: write bf16 C^T as [N,M]   (used to produce Vt)
// 8 waves/block; each wave owns 32x64 (2x4 of 16x16) -> 64x256 block tile.
// ---------------------------------------------------------------------------
template <int MODE>
__global__ __launch_bounds__(256) void gemm_nt_wmma(
    const bf16* __restrict__ A, const float* __restrict__ B,
    float* __restrict__ Cf, bf16* __restrict__ Cb,
    int M, int N, int K) {
  const int lane = threadIdx.x & 31;
  const int wave = threadIdx.x >> 5;
  const int m0 = blockIdx.x * 64  + (wave >> 2) * 32;   // 2 waves along M
  const int n0 = blockIdx.y * 256 + (wave & 3) * 64;    // 4 waves along N

  v8f acc[2][4] = {};

  for (int k0 = 0; k0 < K; k0 += 32) {
    v16bf a0 = load_frag_a(A, m0,      K, k0, lane);
    v16bf a1 = load_frag_a(A, m0 + 16, K, k0, lane);
    v16bf b[4];
#pragma unroll
    for (int j = 0; j < 4; ++j)
      b[j] = load_frag_b_f32(B, n0 + j * 16, K, k0, lane);

    // Unconditional speculative prefetch of the K slab two iterations ahead (global_prefetch_b8)
    __builtin_prefetch(A + (size_t)(m0 + (lane & 15)) * K + k0 + 64, 0, 3);
    __builtin_prefetch(B + (size_t)(n0 + (lane & 15)) * K + k0 + 64, 0, 3);

#pragma unroll
    for (int j = 0; j < 4; ++j) {
      acc[0][j] = wmma_bf16(a0, b[j], acc[0][j]);
      acc[1][j] = wmma_bf16(a1, b[j], acc[1][j]);
    }
  }

  // C/D layout: VGPR r -> lanes 0-15: (M=r, N=lane); lanes 16-31: (M=r+8, N=lane-16)
#pragma unroll
  for (int mi = 0; mi < 2; ++mi) {
#pragma unroll
    for (int ni = 0; ni < 4; ++ni) {
      const int cn    = n0 + ni * 16 + (lane & 15);
      const int rbase = m0 + mi * 16 + ((lane >> 4) << 3);
#pragma unroll
      for (int r = 0; r < 8; ++r) {
        const float val = acc[mi][ni][r];
        const int cm = rbase + r;
        if (MODE == 0)      Cf[(size_t)cm * N + cn] = val;
        else if (MODE == 1) Cb[(size_t)cm * N + cn] = (bf16)val;
        else                Cb[(size_t)cn * M + cm] = (bf16)val;
      }
    }
  }
}

// ---------------------------------------------------------------------------
// Attention: one workgroup per (head, 32-row q tile).
// Q tile (32x128 bf16, 8KB) is staged once into static LDS with CDNA5 async
// loads (global_load_async_to_lds_b128 + s_wait_asynccnt). The full 32x2048
// fp32 score block lives in dynamic LDS (256KB of the WGP's 320KB), so plain
// (non-online) softmax is exact.
//   Q,Km: bf16 [S, DIM]; Vt: bf16 [DIM, S] (V^T); O: bf16 [S, DIM]
// ---------------------------------------------------------------------------
__global__ __launch_bounds__(256) void attn_wmma_kernel(
    const bf16* __restrict__ Q, const bf16* __restrict__ Km,
    const bf16* __restrict__ Vt, bf16* __restrict__ O) {
  extern __shared__ float sc[];                 // [32][SEQ] scores
  __shared__ bf16 Qs[32 * HD];                  // staged Q tile
  const int lane = threadIdx.x & 31;
  const int wave = threadIdx.x >> 5;
  const int q0 = blockIdx.x * 32;
  const int h  = blockIdx.y;
  const float rs = 0.088388347648318447f;       // 1/sqrt(128)

  const bf16* Qh = Q  + (size_t)h * HD;
  const bf16* Kh = Km + (size_t)h * HD;

  // ---- Stage Q tile into LDS via async data mover: 512 b128 chunks, 2/thread
  {
    const unsigned qs_base = (unsigned)(uintptr_t)(&Qs[0]);
#pragma unroll
    for (int c = threadIdx.x; c < 512; c += 256) {
      const int row = c >> 4;                   // 16 x b128 per 128-elem row
      const int col = (c & 15) * 8;
      const bf16* g = Qh + (size_t)(q0 + row) * DIM + col;
      const unsigned loff = qs_base + (unsigned)(row * HD + col) * 2u;
      asm volatile("global_load_async_to_lds_b128 %0, %1, off"
                   :: "v"(loff), "v"(g) : "memory");
    }
    asm volatile("s_wait_asynccnt 0x0" ::: "memory");
  }
  __syncthreads();

  // ---- Phase 1: scores S = Q K^T / sqrt(d). Each wave owns 256 key columns.
  for (int nt = 0; nt < 16; ++nt) {
    const int n0 = wave * 256 + nt * 16;        // key-column base
    v8f s0 = {}, s1 = {};
#pragma unroll
    for (int d0 = 0; d0 < HD; d0 += 32) {
      v16bf a0 = load_frag_a_lds(Qs, 0,  HD, d0, lane);
      v16bf a1 = load_frag_a_lds(Qs, 16, HD, d0, lane);
      v16bf b  = load_frag_b(Kh, n0, DIM, d0, lane);
      s0 = wmma_bf16(a0, b, s0);
      s1 = wmma_bf16(a1, b, s1);
    }
    const int cn = n0 + (lane & 15);
    const int rb = ((lane >> 4) << 3);
#pragma unroll
    for (int r = 0; r < 8; ++r) {
      sc[(rb + r)      * SEQ + cn] = s0[r] * rs;
      sc[(rb + r + 16) * SEQ + cn] = s1[r] * rs;
    }
  }
  __syncthreads();

  // ---- Phase 2: row softmax, 8 threads per row, shfl_xor reductions
  {
    const int row = threadIdx.x >> 3;
    const int sub = threadIdx.x & 7;
    float* rp = sc + row * SEQ;
    float m = -3.4e38f;
    for (int i = sub; i < SEQ; i += 8) m = fmaxf(m, rp[i]);
#pragma unroll
    for (int o = 4; o >= 1; o >>= 1) m = fmaxf(m, __shfl_xor(m, o, 8));
    float sum = 0.f;
    for (int i = sub; i < SEQ; i += 8) {
      const float e = __expf(rp[i] - m);
      rp[i] = e;
      sum += e;
    }
#pragma unroll
    for (int o = 4; o >= 1; o >>= 1) sum += __shfl_xor(sum, o, 8);
    const float inv = 1.f / sum;
    for (int i = sub; i < SEQ; i += 8) rp[i] *= inv;
  }
  __syncthreads();

  // ---- Phase 3: O = P @ V. Each wave owns 16 of the 128 head-dim columns.
  const int nbase = wave * 16;
  v8f o0 = {}, o1 = {};
  for (int k0 = 0; k0 < SEQ; k0 += 32) {
    // A fragments from LDS probabilities (fp32 -> bf16 per lane)
    v16bf a0, a1;
    const int ar   = (lane & 15);
    const int koff = k0 + ((lane >> 4) << 3);
#pragma unroll
    for (int j = 0; j < 8; ++j) {
      a0[j]     = (bf16)sc[ar        * SEQ + koff + j];
      a0[j + 8] = (bf16)sc[ar        * SEQ + koff + 16 + j];
      a1[j]     = (bf16)sc[(ar + 16) * SEQ + koff + j];
      a1[j + 8] = (bf16)sc[(ar + 16) * SEQ + koff + 16 + j];
    }
    // B fragment from Vt: column = head-dim channel, contiguous along k
    v16bf b = load_frag_b(Vt + (size_t)h * HD * SEQ, nbase, SEQ, k0, lane);
    o0 = wmma_bf16(a0, b, o0);
    o1 = wmma_bf16(a1, b, o1);
  }
  const int col = h * HD + nbase + (lane & 15);
  const int rb  = ((lane >> 4) << 3);
#pragma unroll
  for (int r = 0; r < 8; ++r) {
    O[(size_t)(q0 + rb + r)      * DIM + col] = (bf16)o0[r];
    O[(size_t)(q0 + rb + r + 16) * DIM + col] = (bf16)o1[r];
  }
}

// ---------------------------------------------------------------------------
extern "C" void kernel_launch(void* const* d_in, const int* in_sizes, int n_in,
                              void* d_out, int out_size, void* d_ws, size_t ws_size,
                              hipStream_t stream) {
  const float* x  = (const float*)d_in[0];
  const float* wq = (const float*)d_in[1];
  const float* wk = (const float*)d_in[2];
  const float* wv = (const float*)d_in[3];
  const float* wo = (const float*)d_in[4];

  // Workspace layout (bf16): x | Q | K | Vt | O  -> 5 * 16MB = 80MB
  bf16* xb = (bf16*)d_ws;
  bf16* Qb = xb + (size_t)SEQ * DIM;
  bf16* Kb = Qb + (size_t)SEQ * DIM;
  bf16* Vt = Kb + (size_t)SEQ * DIM;
  bf16* Ob = Vt + (size_t)SEQ * DIM;

  f32_to_bf16_kernel<<<2048, 256, 0, stream>>>(x, xb, SEQ * DIM);

  dim3 gproj(SEQ / 64, DIM / 256);
  // Q = x Wq^T, K = x Wk^T (bf16 row-major), Vt = (x Wv^T)^T (bf16 [DIM,S])
  gemm_nt_wmma<1><<<gproj, 256, 0, stream>>>(xb, wq, nullptr, Qb, SEQ, DIM, DIM);
  gemm_nt_wmma<1><<<gproj, 256, 0, stream>>>(xb, wk, nullptr, Kb, SEQ, DIM, DIM);
  gemm_nt_wmma<2><<<gproj, 256, 0, stream>>>(xb, wv, nullptr, Vt, SEQ, DIM, DIM);

  attn_wmma_kernel<<<dim3(SEQ / 32, NH), 256, 32 * SEQ * sizeof(float), stream>>>(
      Qb, Kb, Vt, Ob);

  // out = attn_out @ Wo^T (fp32 output)
  gemm_nt_wmma<0><<<gproj, 256, 0, stream>>>(Ob, wo, (float*)d_out, nullptr,
                                             SEQ, DIM, DIM);
}
